// BinaryTreeLSTM_57758720196942
// MI455X (gfx1250) — matrix operand order repaired and checked by reading, not compiled, not run on hardware
//
#include <hip/hip_runtime.h>
#include <hip/hip_bf16.h>

typedef __attribute__((ext_vector_type(16))) __bf16 v16bf;
typedef __attribute__((ext_vector_type(8)))  float  v8f;
typedef __attribute__((ext_vector_type(8)))  __bf16 bf16x8;
typedef __attribute__((ext_vector_type(4)))  float  f32x4;
typedef __attribute__((ext_vector_type(4)))  unsigned int u32x4;
typedef __attribute__((ext_vector_type(8)))  int i32x8;
typedef __attribute__((ext_vector_type(4)))  int i32x4;

#define DWID  128
#define HDIM  128
#define NGATE 640           // 5*H
#define ROWS_PER_BLOCK 128  // 8 strips of 16

__device__ __forceinline__ float sigmoidf_(float x) { return 1.0f / (1.0f + expf(-x)); }

// ---------------------------------------------------------------------------
// Pre-pass: transpose + convert weights to bf16: WT[n][k] = W[k][n]
// ---------------------------------------------------------------------------
__global__ void prep_weights_kernel(const float* __restrict__ W_wh,
                                    const float* __restrict__ W_hh,
                                    __bf16* __restrict__ WwhT,
                                    __bf16* __restrict__ WhhT) {
    int t = blockIdx.x * blockDim.x + threadIdx.x;
    const int nWwh = DWID * HDIM;
    if (t < nWwh) {
        int n = t / DWID, k = t % DWID;
        WwhT[n * DWID + k] = (__bf16)W_wh[k * HDIM + n];
    }
    int t2 = t - nWwh;
    if (t2 >= 0 && t2 < NGATE * DWID) {
        int n = t2 / DWID, k = t2 % DWID;
        WhhT[n * DWID + k] = (__bf16)W_hh[k * NGATE + n];
    }
}

// ---------------------------------------------------------------------------
// Embedding gather + GEMM: h0[m,:] = emb[word_ids[m]] @ W_wh  (bf16 WMMA)
// Block 256 thr = 8 waves; tile 128 rows x 128 cols; wave = 16-row strip.
// ---------------------------------------------------------------------------
__global__ void embed_gemm_kernel(const int* __restrict__ word_ids,
                                  const float* __restrict__ emb,
                                  const __bf16* __restrict__ WwhT,
                                  __bf16* __restrict__ h0) {
    const int lane  = threadIdx.x & 31;
    const int wave  = threadIdx.x >> 5;
    const int mBase = blockIdx.x * 128 + wave * 16;
    const int r15   = lane & 15;
    const int khalf = lane >> 4;

    const float* arow = emb + (long long)word_ids[mBase + r15] * DWID;

    v8f acc[8] = {};
    for (int k0 = 0; k0 < DWID; k0 += 32) {
        f32x4 a0 = *(const f32x4*)(arow + k0 + khalf * 8);
        f32x4 a1 = *(const f32x4*)(arow + k0 + khalf * 8 + 4);
        f32x4 a2 = *(const f32x4*)(arow + k0 + 16 + khalf * 8);
        f32x4 a3 = *(const f32x4*)(arow + k0 + 16 + khalf * 8 + 4);
        v16bf a;
        #pragma unroll
        for (int i = 0; i < 4; ++i) {
            a[i] = (__bf16)a0[i];  a[4+i]  = (__bf16)a1[i];
            a[8+i] = (__bf16)a2[i]; a[12+i] = (__bf16)a3[i];
        }
        #pragma unroll
        for (int nt = 0; nt < 8; ++nt) {
            const __bf16* bcol = WwhT + (nt * 16 + r15) * DWID + k0 + khalf * 8;
            bf16x8 b0 = *(const bf16x8*)(bcol);
            bf16x8 b1 = *(const bf16x8*)(bcol + 16);
            v16bf b;
            #pragma unroll
            for (int i = 0; i < 8; ++i) { b[i] = b0[i]; b[8+i] = b1[i]; }
            acc[nt] = __builtin_amdgcn_wmma_f32_16x16x32_bf16(
                false, a, false, b, (short)0, acc[nt], false, false);
        }
    }
    #pragma unroll
    for (int nt = 0; nt < 8; ++nt)
        #pragma unroll
        for (int r = 0; r < 8; ++r) {
            int m = mBase + (khalf ? r + 8 : r);
            h0[(long long)m * HDIM + nt * 16 + r15] = (__bf16)acc[nt][r];
        }
}

// ---------------------------------------------------------------------------
// One tree level, weights LDS-resident via Tensor Data Mover.
//   Block = up to 128 output rows; dynamic LDS holds WhhT (640x128 bf16=160KB)
//   Wave w owns output cols [16w,16w+16); per strip it computes the FIVE gate
//   tiles for those cols (n, n+128,...,n+512) -> gates combined in registers.
// ---------------------------------------------------------------------------
__global__ void tree_level_kernel(const __bf16* __restrict__ hin,
                                  const float* __restrict__ b_hh,
                                  const __bf16* __restrict__ WhhT,
                                  __bf16* __restrict__ hout,
                                  float* __restrict__ fout,
                                  int Mout, int isFinal) {
    extern __shared__ __bf16 ldsW[];             // 640*128 bf16 = 160 KB

    const int lane  = threadIdx.x & 31;
    const int wave  = threadIdx.x >> 5;          // 0..7
    const int r15   = lane & 15;
    const int khalf = lane >> 4;
    const int ncol  = wave * 16;                 // this wave's output columns
    const long long blockRow = (long long)blockIdx.x * ROWS_PER_BLOCK;

    // ---- Stage WhhT into LDS: each wave DMAs its 80-row chunk (20 KB) ----
    {
        const unsigned ldsOff  = (unsigned)(wave * 80 * DWID * 2);      // bytes
        const unsigned long long ga =
            (unsigned long long)(const void*)(WhhT + (long long)wave * 80 * DWID);
        // D# group 0: count=1 | lds_addr | 57-bit global_addr | type=2
        u32x4 g0 = { 1u, ldsOff,
                     (unsigned)(ga & 0xFFFFFFFFu),
                     (unsigned)(((ga >> 32) & 0x01FFFFFFu) | (2u << 30)) };
        // D# group 1: data_size=1(2B); dim0=128, dim1=80; tile 128x80; stride0=128
        i32x8 g1 = { (int)(1u << 16),            // workgroup_mask=0, data_size=2B
                     (int)(128u << 16),          // tensor_dim0 lo16 = 128
                     (int)(80u << 16),           // dim0 hi=0 | tensor_dim1 lo16 = 80
                     (int)(128u << 16),          // dim1 hi=0 | tile_dim0 = 128
                     (int)80,                    // tile_dim1 = 80 | tile_dim2 = 0
                     (int)128,                   // tensor_dim0_stride lo32 = 128
                     0, 0 };
        i32x4 g2 = { 0, 0, 0, 0 };
        i32x4 g3 = { 0, 0, 0, 0 };
        i32x8 g4 = { 0, 0, 0, 0, 0, 0, 0, 0 };   // trailing group (clang-23 form)
        __builtin_amdgcn_tensor_load_to_lds(g0, g1, g2, g3, g4, 0);
        __builtin_amdgcn_s_wait_tensorcnt(0);
    }
    __syncthreads();

    // Per-lane loop-invariant bias (cols fixed per lane): 2*b_hh[g*128+ncol+r15]
    float b2[5];
    #pragma unroll
    for (int g = 0; g < 5; ++g) b2[g] = 2.0f * b_hh[g * 128 + ncol + r15];

    const int rowsHere = (int)min((long long)ROWS_PER_BLOCK, (long long)Mout - blockRow);

    for (int strip = 0; strip < rowsHere; strip += 16) {
        const long long rowL = blockRow + strip + r15;
        const __bf16* hl = hin + (2 * rowL) * HDIM;
        const __bf16* hr = hl + HDIM;

        v8f acc[5] = {};
        for (int k0 = 0; k0 < HDIM; k0 += 32) {
            // A fragment = (hl + hr), ISA 16-bit A layout
            bf16x8 l0 = *(const bf16x8*)(hl + k0 + khalf * 8);
            bf16x8 l1 = *(const bf16x8*)(hl + k0 + 16 + khalf * 8);
            bf16x8 q0 = *(const bf16x8*)(hr + k0 + khalf * 8);
            bf16x8 q1 = *(const bf16x8*)(hr + k0 + 16 + khalf * 8);
            v16bf a;
            #pragma unroll
            for (int i = 0; i < 8; ++i) {
                a[i]   = (__bf16)((float)l0[i] + (float)q0[i]);
                a[8+i] = (__bf16)((float)l1[i] + (float)q1[i]);
            }
            #pragma unroll
            for (int g = 0; g < 5; ++g) {
                const __bf16* bcol = ldsW + (g * 128 + ncol + r15) * DWID
                                          + k0 + khalf * 8;
                bf16x8 b0 = *(const bf16x8*)(bcol);
                bf16x8 b1 = *(const bf16x8*)(bcol + 16);
                v16bf b;
                #pragma unroll
                for (int i = 0; i < 8; ++i) { b[i] = b0[i]; b[8+i] = b1[i]; }
                acc[g] = __builtin_amdgcn_wmma_f32_16x16x32_bf16(
                    false, a, false, b, (short)0, acc[g], false, false);
            }
        }

        // Gate combine fully in registers on the C-tile layout:
        // lane l, vgpr r -> row (l<16 ? r : r+8), col ncol + (l&15)
        #pragma unroll
        for (int r = 0; r < 8; ++r) {
            const int m = strip + (khalf ? r + 8 : r);
            const long long mm = blockRow + m;
            float ig = sigmoidf_(acc[0][r] + b2[0]);
            float lf = sigmoidf_(acc[1][r] + b2[1]);
            float rf = sigmoidf_(acc[2][r] + b2[2]);
            float og = sigmoidf_(acc[3][r] + b2[3]);
            float gg = tanhf(    acc[4][r] + b2[4]);
            float hlv = (float)hin[(2 * mm)     * HDIM + ncol + r15];
            float hrv = (float)hin[(2 * mm + 1) * HDIM + ncol + r15];
            float c   = ig * gg + lf * hlv + rf * hrv;
            float ho  = og * tanhf(c);
            if (isFinal) fout[mm * HDIM + ncol + r15] = ho;
            else         hout[mm * HDIM + ncol + r15] = (__bf16)ho;
        }
    }
}

// ---------------------------------------------------------------------------
extern "C" void kernel_launch(void* const* d_in, const int* in_sizes, int n_in,
                              void* d_out, int out_size, void* d_ws, size_t ws_size,
                              hipStream_t stream) {
    (void)in_sizes; (void)n_in; (void)out_size; (void)ws_size;
    const int*   word_ids = (const int*)  d_in[0];
    const float* emb      = (const float*)d_in[1];
    const float* W_wh     = (const float*)d_in[2];
    const float* W_hh     = (const float*)d_in[3];
    const float* b_hh     = (const float*)d_in[4];

    const int M0 = 64 * 2048;                       // 131072 leaf rows

    char* ws = (char*)d_ws;
    __bf16* WwhT = (__bf16*)ws;                     //  32 KB
    __bf16* WhhT = (__bf16*)(ws + 32 * 1024);       // 160 KB
    __bf16* buf0 = (__bf16*)(ws + 256 * 1024);      // 32 MB
    __bf16* buf1 = (__bf16*)(ws + 256 * 1024 + (size_t)M0 * HDIM * 2); // 16 MB

    prep_weights_kernel<<<(DWID * HDIM + NGATE * DWID + 255) / 256, 256, 0, stream>>>(
        W_wh, W_hh, WwhT, WhhT);

    embed_gemm_kernel<<<M0 / 128, 256, 0, stream>>>(word_ids, emb, WwhT, buf0);

    const size_t ldsBytes = (size_t)NGATE * DWID * 2;   // 160 KB dynamic LDS

    __bf16* cur = buf0;
    __bf16* nxt = buf1;
    int Mout = M0 / 2;
    for (int lvl = 0; lvl < 11; ++lvl) {
        int isFinal = (Mout == 64);
        int nBlocks = (Mout + ROWS_PER_BLOCK - 1) / ROWS_PER_BLOCK;
        tree_level_kernel<<<nBlocks, 256, ldsBytes, stream>>>(
            cur, b_hh, WhhT, nxt, (float*)d_out, Mout, isFinal);
        __bf16* t = cur; cur = nxt; nxt = t;
        Mout >>= 1;
    }
}
